// Seq2Seq_50620484550982
// MI455X (gfx1250) — matrix operand order repaired and checked by reading, not compiled, not run on hardware
//
#include <hip/hip_runtime.h>
#include <hip/hip_bf16.h>

// ---------------------------------------------------------------------------
// Seq2Seq LSTM decoder + BERT head + vocab projection for MI455X (gfx1250).
// All GEMMs use v_wmma_f32_16x16x32_bf16 (wave32). fp32 accumulate;
// activations / LayerNorm / c-state fp32.
// Key trick: batch=8 only half fills the WMMA N axis, so the (parallel)
// head and vocab GEMMs pack TWO steps per 16-column tile:
//   columns 0-7  = step 2p  (batch 0-7)
//   columns 8-15 = step 2p+1(batch 0-7)
// -> halves WMMA count and L2 weight traffic of the dominant vocab GEMM.
// ---------------------------------------------------------------------------

#define H      768
#define H4     3072
#define BATCH  8
#define BP     16          // WMMA tile width (2 steps x 8 batch, or batch pad)
#define STEPS  511
#define NPAIR  256         // ceil(STEPS / 2) packed step-pairs
#define VOC    21128
#define VOCP   21248       // 83 blocks * 8 waves * 2 tiles * 16 rows
#define LN_EPS 1e-12f

#define ENABLE_ASYNC_STAGE 1   // global_load_async_to_lds_b128 staging path

typedef __bf16 bf16_t;
typedef __attribute__((ext_vector_type(16))) __bf16 v16bf;
typedef __attribute__((ext_vector_type(8)))  __bf16 v8bf;
typedef __attribute__((ext_vector_type(8)))  float  v8f;

__device__ inline v8f wmma_bf16(v16bf a, v16bf b, v8f c) {
  return __builtin_amdgcn_wmma_f32_16x16x32_bf16(false, a, false, b, (short)0, c,
                                                 false, false);
}

// A-operand (16x32, 16-bit) per-lane load, contiguous-K layout (B uses the same):
// lanes 0-15  : M=lane,    K = [kA..kA+15]   (kA = k*32)
// lanes 16-31 : M=lane-16, same but kA = k*32+16
// Caller passes p = &W[row*H + kA]; we fetch halves [0..7] and [8..15].
__device__ inline v16bf ldA(const bf16_t* __restrict__ p) {
  v8bf lo = *(const v8bf*)(p);
  v8bf hi = *(const v8bf*)(p + 8);
  return __builtin_shufflevector(lo, hi, 0, 1, 2, 3, 4, 5, 6, 7,
                                         8, 9, 10, 11, 12, 13, 14, 15);
}

__device__ inline float sigm(float x) { return 1.0f / (1.0f + __expf(-x)); }

// Stage one 16xH bf16 tile (24KB) from global into LDS.
// Async path: GLOBAL_LOAD_ASYNC_TO_LDS_B128 writes LDS directly (no VGPR
// round trip), tracked by ASYNCcnt; each wave waits its own asyncs before the
// caller's barrier. LDS operand = low 32 bits of the flat shared address
// (ISA: flat LDS addresses truncate to the workgroup-relative byte offset).
__device__ inline void stage_tile(const bf16_t* __restrict__ src,
                                  bf16_t* dst_lds, int tid) {
#if ENABLE_ASYNC_STAGE
#pragma unroll
  for (int i = 0; i < (BP * H) / (256 * 8); ++i) {   // 6 x 16B per thread
    const int e = (i * 256 + tid) * 8;
    asm volatile("global_load_async_to_lds_b128 %0, %1, off"
                 :
                 : "v"((unsigned)(unsigned long long)(dst_lds + e)),
                   "v"(src + e)
                 : "memory");
  }
  asm volatile("s_wait_asynccnt 0x0" ::: "memory");
#else
  const uint4* s = (const uint4*)src;
  uint4* d = (uint4*)dst_lds;
  for (int i = tid; i < BP * H / 8; i += 256) d[i] = s[i];
#endif
}

// ---------------------------------------------------------------------------
// Prep kernels: one-time fp32 -> bf16 weight conversion (weights then stay
// L2-resident; total resident set ~62MB << 192MB L2).
// ---------------------------------------------------------------------------
__global__ void k_prep_w(const float* __restrict__ Wih, const float* __restrict__ Whh,
                         bf16_t* __restrict__ Wih_b, bf16_t* __restrict__ Wsum_b) {
  size_t i = (size_t)blockIdx.x * 256 + threadIdx.x;
  if (i < (size_t)H4 * H) {
    float a = Wih[i], b = Whh[i];
    Wih_b[i]  = (bf16_t)a;
    Wsum_b[i] = (bf16_t)(a + b);   // x == h for steps >= 1, fold the two matmuls
  }
}

__global__ void k_prep_wd(const float* __restrict__ Wd, bf16_t* __restrict__ Wd_b) {
  size_t i = (size_t)blockIdx.x * 256 + threadIdx.x;
  if (i < (size_t)H * H) Wd_b[i] = (bf16_t)Wd[i];
}

__global__ void k_prep_wv(const float* __restrict__ Wv, bf16_t* __restrict__ Wv_b) {
  size_t i = (size_t)blockIdx.x * 256 + threadIdx.x;
  if (i < (size_t)VOCP * H) {
    size_t r = i / H;
    if (r < VOC) Wv_b[i] = (bf16_t)Wv[i];
    else         Wv_b[i] = (bf16_t)0.0f;     // zero pad rows -> zero logits
  }
}

__global__ void k_prep_misc(const float* __restrict__ bih, const float* __restrict__ bhh,
                            const float* __restrict__ cls,
                            float* __restrict__ bsum, bf16_t* __restrict__ x0) {
  int i = blockIdx.x * 256 + threadIdx.x;
  if (i < H4) bsum[i] = bih[i] + bhh[i];
  if (i < BP * H) {
    int b = i / H;
    x0[i] = (b < BATCH) ? (bf16_t)cls[i] : (bf16_t)0.0f;   // pad batch rows = 0
  }
}

// ---------------------------------------------------------------------------
// Kernel A: sequential LSTM chain. One workgroup (16 waves). State in LDS:
// xh = current h (bf16, B-operand layout = row-major 16xH), cbuf = c (f32).
// Each wave owns 3 hidden-blocks of 16; i/f/g/o tile accumulators stay in
// registers (identical D layouts -> elementwise gate fusion, no LDS round
// trip for gates). h_t is emitted into the PACKED pair layout:
//   th[pair][ (t-1)%2 * 8 + batch ][H]
// ---------------------------------------------------------------------------
#define LSTM_THREADS 512
#define LSTM_WAVES   16

__global__ __launch_bounds__(LSTM_THREADS)
void k_lstm(const bf16_t* __restrict__ Wih, const bf16_t* __restrict__ Wsum,
            const float* __restrict__ bsum, const bf16_t* __restrict__ x0,
            bf16_t* __restrict__ th) {
  __shared__ __align__(32) bf16_t xh[BP * H];     // 24 KB
  __shared__ float cbuf[BATCH * H];               // 24 KB

  const int tid = threadIdx.x, wave = tid >> 5, lane = tid & 31;

  for (int i = tid; i < BP * H; i += LSTM_THREADS) xh[i] = x0[i];
  for (int i = tid; i < BATCH * H; i += LSTM_THREADS) cbuf[i] = 0.0f;
  {  // zero the unused odd-tail rows of the last pair tile (step 511 absent)
    const size_t base = ((size_t)(NPAIR - 1) * BP + 8) * H;
    for (int i = tid; i < 8 * H; i += LSTM_THREADS) th[base + i] = (bf16_t)0.0f;
  }
  __syncthreads();

  const int n    = lane & 15;            // batch column (N)
  const int rA   = lane & 15;            // A row within tile
  const int kAo  = (lane >> 4) << 4;     // A K-offset: +0 / +16
  const int kBo  = (lane >> 4) << 4;     // B K-offset: +0 / +16
  const int moff = (lane >> 4) << 3;     // D row offset: +0 / +8

  for (int t = 0; t <= STEPS; ++t) {
    const bf16_t* W = (t == 0) ? Wih : Wsum;   // first cell: x=cls, h=0
    v8f hres[3];

#pragma unroll
    for (int r = 0; r < 3; ++r) {
      const int jb = wave + LSTM_WAVES * r;    // hidden block (48 total)
      v8f a0 = {}, a1 = {}, a2 = {}, a3 = {};  // i, f, g, o tiles
      const bf16_t* p0 = W + (size_t)(0 * H + jb * 16 + rA) * H + kAo;
      const bf16_t* p1 = W + (size_t)(1 * H + jb * 16 + rA) * H + kAo;
      const bf16_t* p2 = W + (size_t)(2 * H + jb * 16 + rA) * H + kAo;
      const bf16_t* p3 = W + (size_t)(3 * H + jb * 16 + rA) * H + kAo;
      for (int k = 0; k < H / 32; ++k) {
        v16bf b = *(const v16bf*)(&xh[n * H + k * 32 + kBo]);
        a0 = wmma_bf16(ldA(p0 + k * 32), b, a0);
        a1 = wmma_bf16(ldA(p1 + k * 32), b, a1);
        a2 = wmma_bf16(ldA(p2 + k * 32), b, a2);
        a3 = wmma_bf16(ldA(p3 + k * 32), b, a3);
      }
      const int jbase = jb * 16 + moff;
#pragma unroll
      for (int i = 0; i < 8; ++i) {
        const int j = jbase + i;
        float iv = sigm(a0[i] + bsum[0 * H + j]);
        float fv = sigm(a1[i] + bsum[1 * H + j]);
        float gv = tanhf(a2[i] + bsum[2 * H + j]);
        float ov = sigm(a3[i] + bsum[3 * H + j]);
        float cnew;
        if (n < BATCH) {
          float cold = cbuf[n * H + j];     // same thread owns (n,j) every step
          cnew = fv * cold + iv * gv;
          cbuf[n * H + j] = cnew;
        } else {
          cnew = iv * gv;
        }
        hres[r][i] = ov * tanhf(cnew);
      }
    }
    __syncthreads();                      // everyone done reading old xh
    if (n < BATCH) {
      const int idx  = t - 1;                 // emitted step index
      const int pair = idx >> 1;
      const int prow = ((idx & 1) << 3) + n;  // packed row in pair tile
#pragma unroll
      for (int r = 0; r < 3; ++r) {
        const int jbase = (wave + LSTM_WAVES * r) * 16 + moff;
#pragma unroll
        for (int i = 0; i < 8; ++i) {
          bf16_t hv = (bf16_t)hres[r][i];
          xh[n * H + jbase + i] = hv;
          if (t > 0) th[((size_t)pair * BP + prow) * H + jbase + i] = hv;
        }
      }
    }
    __syncthreads();                      // new xh visible before next step
  }
}

// ---------------------------------------------------------------------------
// Kernel B: head transform, one block (8 waves) per STEP-PAIR, fully parallel.
// dense(HxH) via WMMA (N = 2 steps x 8 batch) -> +bias -> exact gelu ->
// LayerNorm (wave-shuffle, 2 rows per wave) -> bf16 packed state for vocab.
// ---------------------------------------------------------------------------
__global__ __launch_bounds__(256)
void k_head(const bf16_t* __restrict__ th, const bf16_t* __restrict__ Wd,
            const float* __restrict__ bd, const float* __restrict__ gamma,
            const float* __restrict__ beta, bf16_t* __restrict__ tt) {
  __shared__ float tbuf[BP * H];                  // 48 KB (16 rows)

  const int pair = blockIdx.x;
  const int tid = threadIdx.x, wave = tid >> 5, lane = tid & 31;
  const bf16_t* T = th + (size_t)pair * BP * H;

  const int n    = lane & 15;
  const int rA   = lane & 15;
  const int kAo  = (lane >> 4) << 4;
  const int kBo  = (lane >> 4) << 4;
  const int moff = (lane >> 4) << 3;

#pragma unroll
  for (int r = 0; r < 6; ++r) {                   // 48 blocks / 8 waves
    const int jb = wave + 8 * r;
    v8f acc = {};
    const bf16_t* pa = Wd + (size_t)(jb * 16 + rA) * H + kAo;
    for (int k = 0; k < H / 32; ++k) {
      v16bf b = *(const v16bf*)(T + n * H + k * 32 + kBo);
      acc = wmma_bf16(ldA(pa + k * 32), b, acc);
    }
#pragma unroll
    for (int i = 0; i < 8; ++i) {                 // all 16 columns are live
      const int j = jb * 16 + moff + i;
      float x = acc[i] + bd[j];
      tbuf[n * H + j] = 0.5f * x * (1.0f + erff(x * 0.70710678118654752f));
    }
  }
  __syncthreads();

  // LayerNorm: 16 rows (2 steps x 8 batch), 2 rows per wave.
#pragma unroll
  for (int rr = 0; rr < 2; ++rr) {
    const int row = wave + 8 * rr;
    float s = 0.0f, s2 = 0.0f;
    for (int j = lane; j < H; j += 32) {
      float v = tbuf[row * H + j];
      s += v;
      s2 += v * v;
    }
    for (int off = 16; off > 0; off >>= 1) {
      s  += __shfl_xor(s, off, 32);
      s2 += __shfl_xor(s2, off, 32);
    }
    const float mu = s * (1.0f / H);
    const float var = s2 * (1.0f / H) - mu * mu;
    const float rstd = rsqrtf(var + LN_EPS);
    bf16_t* outr = tt + ((size_t)pair * BP + row) * H;
    for (int j = lane; j < H; j += 32) {
      float v = (tbuf[row * H + j] - mu) * rstd * gamma[j] + beta[j];
      outr[j] = (bf16_t)v;
    }
  }
}

// ---------------------------------------------------------------------------
// Kernel C: vocab projection (the FLOP workhorse, W_voc streamed from L2).
// Grid (83, pairs); each block async-stages the pair's 16xH bf16 state into
// LDS once, then 8 waves each produce 2x16 vocab rows (one B load feeds two
// WMMAs, all 16 N columns live = 2 steps). float4 stores, step-guarded demux.
// ---------------------------------------------------------------------------
__global__ __launch_bounds__(256)
void k_vocab(const bf16_t* __restrict__ tt, const bf16_t* __restrict__ Wv,
             const float* __restrict__ bv, float* __restrict__ out) {
  __shared__ __align__(32) bf16_t tsh[BP * H];    // 24 KB

  const int pair = blockIdx.y;
  const int cb   = blockIdx.x;                    // 256 vocab rows per block
  const int tid = threadIdx.x, wave = tid >> 5, lane = tid & 31;

  stage_tile(tt + (size_t)pair * BP * H, tsh, tid);
  __syncthreads();

  const int n    = lane & 15;
  const int rA   = lane & 15;
  const int kAo  = (lane >> 4) << 4;
  const int kBo  = (lane >> 4) << 4;
  const int moff = (lane >> 4) << 3;

  const int v0 = cb * 256 + wave * 32;            // two 16-row tiles per wave
  v8f acc0 = {}, acc1 = {};
  const bf16_t* pa0 = Wv + (size_t)(v0 + rA) * H + kAo;
  const bf16_t* pa1 = Wv + (size_t)(v0 + 16 + rA) * H + kAo;
  for (int k = 0; k < H / 32; ++k) {
    if (k + 2 < H / 32) {                          // stream W_voc rows from L2
      __builtin_prefetch(pa0 + (k + 2) * 32, 0, 0);
      __builtin_prefetch(pa1 + (k + 2) * 32, 0, 0);
    }
    v16bf b = *(const v16bf*)(&tsh[n * H + k * 32 + kBo]);
    acc0 = wmma_bf16(ldA(pa0 + k * 32), b, acc0);
    acc1 = wmma_bf16(ldA(pa1 + k * 32), b, acc1);
  }

  const int sstep = 2 * pair + (n >> 3);          // demux packed column
  const int b     = n & 7;
  if (sstep < STEPS) {
    float* orow = out + ((size_t)sstep * BATCH + b) * VOC;
#pragma unroll
    for (int tile = 0; tile < 2; ++tile) {
      const v8f& acc = tile ? acc1 : acc0;
      const int vb = v0 + tile * 16 + moff;
      if (vb + 8 <= VOC) {
        float4 lo = make_float4(acc[0] + bv[vb + 0], acc[1] + bv[vb + 1],
                                acc[2] + bv[vb + 2], acc[3] + bv[vb + 3]);
        float4 hi = make_float4(acc[4] + bv[vb + 4], acc[5] + bv[vb + 5],
                                acc[6] + bv[vb + 6], acc[7] + bv[vb + 7]);
        *(float4*)(orow + vb)     = lo;
        *(float4*)(orow + vb + 4) = hi;
      } else {
#pragma unroll
        for (int i = 0; i < 8; ++i)
          if (vb + i < VOC) orow[vb + i] = acc[i] + bv[vb + i];
      }
    }
  }
}

// ---------------------------------------------------------------------------
extern "C" void kernel_launch(void* const* d_in, const int* in_sizes, int n_in,
                              void* d_out, int out_size, void* d_ws, size_t ws_size,
                              hipStream_t stream) {
  const float* cls  = (const float*)d_in[0];
  const float* W_ih = (const float*)d_in[1];
  const float* W_hh = (const float*)d_in[2];
  const float* b_ih = (const float*)d_in[3];
  const float* b_hh = (const float*)d_in[4];
  const float* W_d  = (const float*)d_in[5];
  const float* b_d  = (const float*)d_in[6];
  const float* gam  = (const float*)d_in[7];
  const float* bet  = (const float*)d_in[8];
  const float* W_v  = (const float*)d_in[9];
  const float* b_v  = (const float*)d_in[10];
  // d_in[11] = steps (compile-time constant STEPS here)
  float* out = (float*)d_out;

  char* ws = (char*)d_ws;
  size_t off = 0;
  auto alloc = [&](size_t bytes) -> void* {
    void* p = ws + off;
    off += (bytes + 255) & ~(size_t)255;
    return p;
  };
  bf16_t* Wih_b  = (bf16_t*)alloc((size_t)H4 * H * 2);
  bf16_t* Wsum_b = (bf16_t*)alloc((size_t)H4 * H * 2);
  bf16_t* Wd_b   = (bf16_t*)alloc((size_t)H * H * 2);
  bf16_t* Wv_b   = (bf16_t*)alloc((size_t)VOCP * H * 2);
  float*  bsum   = (float*)alloc((size_t)H4 * 4);
  bf16_t* x0     = (bf16_t*)alloc((size_t)BP * H * 2);
  bf16_t* th     = (bf16_t*)alloc((size_t)NPAIR * BP * H * 2);
  bf16_t* tt     = (bf16_t*)alloc((size_t)NPAIR * BP * H * 2);

  k_prep_w<<<(H4 * H + 255) / 256, 256, 0, stream>>>(W_ih, W_hh, Wih_b, Wsum_b);
  k_prep_wd<<<(H * H + 255) / 256, 256, 0, stream>>>(W_d, Wd_b);
  k_prep_wv<<<(int)(((size_t)VOCP * H + 255) / 256), 256, 0, stream>>>(W_v, Wv_b);
  k_prep_misc<<<(BP * H + 255) / 256, 256, 0, stream>>>(b_ih, b_hh, cls, bsum, x0);

  k_lstm<<<1, LSTM_THREADS, 0, stream>>>(Wih_b, Wsum_b, bsum, x0, th);
  k_head<<<NPAIR, 256, 0, stream>>>(th, Wd_b, b_d, gam, bet, tt);
  k_vocab<<<dim3(83, NPAIR), 256, 0, stream>>>(tt, Wv_b, b_v, out);
}
